// GroupedQueryAttention_35158602285711
// MI455X (gfx1250) — matrix operand order repaired and checked by reading, not compiled, not run on hardware
//
#include <hip/hip_runtime.h>
#include <hip/hip_bf16.h>

// ---------------------------------------------------------------------------
// GQA forward for gfx1250 (MI455X): f16 WMMA GEMMs + flash attention.
// B=2, S=2048, D=2048, H=16, KV=4, HD=128, GROUP=4, SCALING=1/16.
// Double-buffered LDS tiles; global->LDS staging uses the CDNA5 async-to-LDS
// path (GLOBAL_LOAD_ASYNC_TO_LDS_B128, ASYNCcnt) via inline asm.
// ---------------------------------------------------------------------------

typedef __attribute__((ext_vector_type(16))) _Float16 v16h;
typedef __attribute__((ext_vector_type(8)))  float    v8f;

#define SB      2048
#define SD      2048
#define NH      16
#define NKV     4
#define HD      128
#define SCALING 0.0625f

// 16-byte async copy global->LDS (ISA §15.18.3 op 98). VDST = LDS byte
// address (low 32 bits of the generic shared pointer, per flat-aperture
// rules), VADDR = 64-bit global address, SADDR = off (NULL).
__device__ __forceinline__ void async_copy16(const void* g, void* l) {
  unsigned lds = (unsigned)(unsigned long long)l;
  asm volatile("global_load_async_to_lds_b128 %0, %1, off"
               :
               : "v"(lds), "v"(g)
               : "memory");
}

__device__ __forceinline__ void async_wait0() {
#if __has_builtin(__builtin_amdgcn_s_wait_asynccnt)
  __builtin_amdgcn_s_wait_asynccnt(0);
#else
  asm volatile("s_wait_asynccnt 0x0" ::: "memory");
#endif
}

// ---- WMMA helper -----------------------------------------------------------
__device__ __forceinline__ v8f wmma16(v16h a, v16h b, v8f c) {
  return __builtin_amdgcn_wmma_f32_16x16x32_f16(
      /*neg_a=*/false, a, /*neg_b=*/false, b,
      /*c_mod=*/(short)0, c, /*reuse_a=*/false, /*reuse_b=*/false);
}

// A-fragment (16x32 f16): lane lr(0..15)=row, lh selects K half; contiguous K
// (k = 16*lh + s). Pairs contiguous -> compiler merges to 2x ds_load_b128.
__device__ __forceinline__ v16h load_a16(const _Float16* p, int ld, int lr, int lh) {
  v16h a;
  const _Float16* rp = p + lr * ld + lh * 16;
#pragma unroll
  for (int pj = 0; pj < 8; ++pj) {
    int kb = pj * 2;
    union { unsigned int u; _Float16 h[2]; } cv;
    cv.u = *reinterpret_cast<const unsigned int*>(rp + kb);
    a[2 * pj]     = cv.h[0];
    a[2 * pj + 1] = cv.h[1];
  }
  return a;
}

// B-fragment with contiguous K: p points at this lane's 16 consecutive halves.
__device__ __forceinline__ v16h load_bt16(const _Float16* p) {
  v16h b;
#pragma unroll
  for (int pj = 0; pj < 8; ++pj) {
    union { unsigned int u; _Float16 h[2]; } cv;
    cv.u = *reinterpret_cast<const unsigned int*>(p + 2 * pj);
    b[2 * pj]     = cv.h[0];
    b[2 * pj + 1] = cv.h[1];
  }
  return b;
}

// Transpose-store one (2-row x 8-col) register chunk into a K-major LDS tile
// (dst[col*32 + k]); k pairs packed as u32 stores.
__device__ __forceinline__ void transpose_store_pair(_Float16* dst, int4 r0, int4 r1,
                                                     int kp, int cg) {
  union { int4 v; _Float16 h[8]; } ua, ub;
  ua.v = r0;
  ub.v = r1;
#pragma unroll
  for (int c = 0; c < 8; ++c) {
    union { unsigned int u; _Float16 h[2]; } cv;
    cv.h[0] = ua.h[c];
    cv.h[1] = ub.h[c];
    *reinterpret_cast<unsigned int*>(&dst[(cg * 8 + c) * 32 + 2 * kp]) = cv.u;
  }
}

// ---- fp32 -> f16 convert ---------------------------------------------------
__global__ void cvt_f32_f16(const float* __restrict__ in, _Float16* __restrict__ out,
                            size_t n) {
  size_t i = (size_t)blockIdx.x * blockDim.x + threadIdx.x;
  size_t stride = (size_t)gridDim.x * blockDim.x;
  for (; i < n; i += stride) out[i] = (_Float16)in[i];
}

// ---- Generic f16 GEMM: C[M,N] (f32) = A[M,K] @ B[K,N], 128x128 tile --------
__global__ __launch_bounds__(256)
void gemm_f16f32(const _Float16* __restrict__ A, const _Float16* __restrict__ Bm,
                 float* __restrict__ C, int M, int N, int K) {
  __shared__ __align__(16) _Float16 As[2][128 * 32];  // row-major [m][k]
  __shared__ __align__(16) _Float16 Bs[2][128 * 32];  // K-major  [n][k]

  const int t = threadIdx.x;
  const int lane = t & 31, w = t >> 5;
  const int lr = lane & 15, lh = lane >> 4;
  const int wm = w >> 2, wn = w & 3;        // waves: 2 (M) x 4 (N)
  const int mBase = blockIdx.y * 128;
  const int nBase = blockIdx.x * 128;
  const int kp = t >> 4;                    // 0..15 -> k pair 2*kp
  const int cg = t & 15;                    // 0..15 -> 8-column group

  // Per-thread staging addresses (advance by constants per K step).
  const int aRow = t >> 1, aSeg = (t & 1) * 16;
  const _Float16* aG = A + (size_t)(mBase + aRow) * K + aSeg;
  const _Float16* bG = Bm + (size_t)(2 * kp) * N + nBase + cg * 8;
  const int aOff = aRow * 32 + aSeg;

  v8f acc[4][2];
#pragma unroll
  for (int mt = 0; mt < 4; ++mt)
#pragma unroll
    for (int nt = 0; nt < 2; ++nt) acc[mt][nt] = {};

  // ---- stage tile 0 into buffer 0 (A async, B register-transposed)
  async_copy16(aG, &As[0][aOff]);
  async_copy16(aG + 8, &As[0][aOff + 8]);
  transpose_store_pair(&Bs[0][0],
                       *reinterpret_cast<const int4*>(bG),
                       *reinterpret_cast<const int4*>(bG + N), kp, cg);
  async_wait0();
  __syncthreads();

  int cur = 0;
  for (int k0 = 0; k0 < K; k0 += 32) {
    const bool more = (k0 + 32) < K;
    int4 b0n = {}, b1n = {};
    if (more) {
      aG += 32;
      bG += (size_t)32 * N;
      async_copy16(aG, &As[cur ^ 1][aOff]);
      async_copy16(aG + 8, &As[cur ^ 1][aOff + 8]);
      b0n = *reinterpret_cast<const int4*>(bG);
      b1n = *reinterpret_cast<const int4*>(bG + N);
    }

    // compute on buffer `cur`
    v16h aF[4];
#pragma unroll
    for (int mt = 0; mt < 4; ++mt)
      aF[mt] = load_a16(&As[cur][(wm * 64 + mt * 16) * 32], 32, lr, lh);
    v16h bF[2];
#pragma unroll
    for (int nt = 0; nt < 2; ++nt)
      bF[nt] = load_bt16(&Bs[cur][(wn * 32 + nt * 16 + lr) * 32 + lh * 16]);

#pragma unroll
    for (int mt = 0; mt < 4; ++mt)
#pragma unroll
      for (int nt = 0; nt < 2; ++nt)
        acc[mt][nt] = wmma16(aF[mt], bF[nt], acc[mt][nt]);

    if (more) {
      transpose_store_pair(&Bs[cur ^ 1][0], b0n, b1n, kp, cg);
      async_wait0();
    }
    __syncthreads();
    cur ^= 1;
  }

#pragma unroll
  for (int mt = 0; mt < 4; ++mt)
#pragma unroll
    for (int nt = 0; nt < 2; ++nt) {
      int col = nBase + wn * 32 + nt * 16 + lr;
#pragma unroll
      for (int r = 0; r < 8; ++r) {
        int row = mBase + wm * 64 + mt * 16 + lh * 8 + r;
        C[(size_t)row * N + col] = acc[mt][nt][r];
      }
    }
}

// ---- RMSNorm + RoPE + scale, [B*S][nh*128] f32 -> [B][nh][S][128] f16 ------
__global__ __launch_bounds__(128)
void qk_norm_rope(const float* __restrict__ raw, const float* __restrict__ scale,
                  const float* __restrict__ cosb, const float* __restrict__ sinb,
                  _Float16* __restrict__ outp, int nh, float outScale) {
  int idx = blockIdx.x;
  int h = idx % nh;
  int s = (idx / nh) % SB;
  int b = idx / (nh * SB);
  int d = threadIdx.x;

  const float* rp = raw + ((size_t)(b * SB + s) * nh + h) * HD;
  float xv = rp[d];

  __shared__ float red[128];
  red[d] = xv * xv;
  __syncthreads();
#pragma unroll
  for (int off = 64; off > 0; off >>= 1) {
    if (d < off) red[d] += red[d + off];
    __syncthreads();
  }
  float inv = rsqrtf(red[0] * (1.0f / HD) + 1e-6f);

  float xn = xv * inv * scale[d];
  int dp = (d < 64) ? d + 64 : d - 64;
  float part = rp[dp] * inv * scale[dp];
  float rot = (d < 64) ? -part : part;
  float ov = xn * cosb[s * HD + d] + rot * sinb[s * HD + d];
  outp[(((size_t)(b * nh + h)) * SB + s) * HD + d] = (_Float16)(ov * outScale);
}

// ---- V: [B*S][KV*128] f32 -> [B][KV][S][128] f16 ---------------------------
__global__ void v_transpose(const float* __restrict__ raw, _Float16* __restrict__ outp) {
  size_t n = (size_t)2 * SB * NKV * HD;
  size_t stride = (size_t)gridDim.x * blockDim.x;
  for (size_t i = (size_t)blockIdx.x * blockDim.x + threadIdx.x; i < n; i += stride) {
    int d   = (int)(i & 127);
    int kvh = (int)((i >> 7) & 3);
    int s   = (int)((i >> 9) & 2047);
    int b   = (int)(i >> 20);
    outp[(((size_t)(b * NKV + kvh)) * SB + s) * HD + d] = (_Float16)raw[i];
  }
}

// ---- Flash attention: one block per (b, h, 128 q-rows) ---------------------
// LDS plan (40 KB): smem[16384] f16 holds Q (128x128) during fragment build,
// then is reused as double-buffered K (row-major, offsets 0/4096) and V
// (K-major, offsets 8192/12288) tiles. Buffer addresses are computed
// arithmetically (no local pointer arrays -> no addrspacecast initializers).
__global__ __launch_bounds__(256)
void attn_flash(const _Float16* __restrict__ qh, const _Float16* __restrict__ kh,
                const _Float16* __restrict__ vh, _Float16* __restrict__ ctxh) {
  __shared__ __align__(16) _Float16 smem[16384];     // 32 KB
  __shared__ __align__(16) _Float16 Ps[8][16 * 32];  //  8 KB (wave-private)

  const int t = threadIdx.x, lane = t & 31, w = t >> 5;
  const int lr = lane & 15, lh = lane >> 4;
  const int bh = blockIdx.x;          // b*NH + h
  const int b = bh / NH, h = bh % NH;
  const int kvh = h / (NH / NKV);
  const int qBase = blockIdx.y * 128;
  const int kp = t >> 4, cg = t & 15;

  // ---- Load the 128x128 Q block (async) and build persistent A-fragments.
  const _Float16* qg = qh + (((size_t)(b * NH + h)) * SB + qBase) * HD;
#pragma unroll
  for (int j = 0; j < 8; ++j) {
    int i = t + j * 256;
    int row = i >> 4, seg = i & 15;
    async_copy16(qg + (size_t)row * HD + seg * 8, &smem[row * 128 + seg * 8]);
  }
  async_wait0();
  __syncthreads();

  v16h qF[4];
#pragma unroll
  for (int ks = 0; ks < 4; ++ks)
    qF[ks] = load_a16(&smem[(w * 16) * 128 + ks * 32], 128, lr, lh);
  __syncthreads();   // Q reads done before smem is reused for K/V

  v8f ctxAcc[8];
#pragma unroll
  for (int d = 0; d < 8; ++d) ctxAcc[d] = {};
  float mrow[8], lrow[8];
#pragma unroll
  for (int r = 0; r < 8; ++r) { mrow[r] = -1e30f; lrow[r] = 0.0f; }

  const _Float16* kg = kh + ((size_t)(b * NKV + kvh)) * SB * HD;
  const _Float16* vg = vh + ((size_t)(b * NKV + kvh)) * SB * HD;

  // Per-thread staging addresses for K (2 x b128/thread, row-major, async)
  // and V (2x8 chunk/thread, transposed to K-major via registers).
  const int kRow = t >> 4, kSeg = (t & 15) * 8;
  const int kOff = kRow * 128 + kSeg;
  const _Float16* kG = kg + (size_t)kRow * HD + kSeg;
  const _Float16* vG = vg + (size_t)(2 * kp) * HD + cg * 8;

  // ---- stage kt=0 into buffer 0
  async_copy16(kG, &smem[kOff]);
  async_copy16(kG + 16 * HD, &smem[kOff + 16 * 128]);
  transpose_store_pair(&smem[8192],
                       *reinterpret_cast<const int4*>(vG),
                       *reinterpret_cast<const int4*>(vG + HD), kp, cg);
  async_wait0();
  __syncthreads();

  const int ktMax = qBase / 32 + 4;   // causal: tiles up to qBase+128
  int cur = 0;
  for (int kt = 0; kt < ktMax; ++kt) {
    const bool more = (kt + 1) < ktMax;
    const int nxtOff = (cur ^ 1) * 4096;
    int4 v0n = {}, v1n = {};
    if (more) {
      const _Float16* kGn = kG + (size_t)(kt + 1) * 32 * HD;
      const _Float16* vGn = vG + (size_t)(kt + 1) * 32 * HD;
      async_copy16(kGn, &smem[nxtOff + kOff]);
      async_copy16(kGn + 16 * HD, &smem[nxtOff + kOff + 16 * 128]);
      v0n = *reinterpret_cast<const int4*>(vGn);
      v1n = *reinterpret_cast<const int4*>(vGn + HD);
    }

    const _Float16* Ks = smem + cur * 4096;          // row-major [k][d]
    const _Float16* Vs = smem + 8192 + cur * 4096;   // K-major  [d][k]

    // scores tile: 16 q-rows x 32 k-cols per wave, f32 accum.
    v8f s0 = {}, s1 = {};
#pragma unroll
    for (int ks = 0; ks < 4; ++ks) {
      v16h b0 = load_bt16(&Ks[(0 * 16 + lr) * 128 + ks * 32 + lh * 16]);
      v16h b1 = load_bt16(&Ks[(1 * 16 + lr) * 128 + ks * 32 + lh * 16]);
      s0 = wmma16(qF[ks], b0, s0);
      s1 = wmma16(qF[ks], b1, s1);
    }

    // online softmax (rows in vgpr slots; cols across 16-lane groups)
    const int colB = kt * 32;
#pragma unroll
    for (int r = 0; r < 8; ++r) {
      int rowg = qBase + w * 16 + lh * 8 + r;
      float v0 = s0[r]; if (colB + lr > rowg)      v0 = -1e30f;
      float v1 = s1[r]; if (colB + 16 + lr > rowg) v1 = -1e30f;
      float tmax = fmaxf(v0, v1);
      tmax = fmaxf(tmax, __shfl_xor(tmax, 1));
      tmax = fmaxf(tmax, __shfl_xor(tmax, 2));
      tmax = fmaxf(tmax, __shfl_xor(tmax, 4));
      tmax = fmaxf(tmax, __shfl_xor(tmax, 8));
      float mn = fmaxf(mrow[r], tmax);
      float al = __expf(mrow[r] - mn);
      float p0 = __expf(v0 - mn);
      float p1 = __expf(v1 - mn);
      float rs = p0 + p1;
      rs += __shfl_xor(rs, 1);
      rs += __shfl_xor(rs, 2);
      rs += __shfl_xor(rs, 4);
      rs += __shfl_xor(rs, 8);
      lrow[r] = lrow[r] * al + rs;
      mrow[r] = mn;
      Ps[w][(lh * 8 + r) * 32 + lr]      = (_Float16)p0;
      Ps[w][(lh * 8 + r) * 32 + 16 + lr] = (_Float16)p1;
#pragma unroll
      for (int d = 0; d < 8; ++d) ctxAcc[d][r] *= al;
    }

    // PV: P(16x32) @ V(32x128) via 8 WMMAs; V fragments are contiguous-K.
    v16h pF = load_a16(&Ps[w][0], 32, lr, lh);
#pragma unroll
    for (int d = 0; d < 8; ++d) {
      v16h vF = load_bt16(&Vs[(d * 16 + lr) * 32 + lh * 16]);
      ctxAcc[d] = wmma16(pF, vF, ctxAcc[d]);
    }

    if (more) {
      transpose_store_pair(&smem[8192 + nxtOff], v0n, v1n, kp, cg);
      async_wait0();
    }
    __syncthreads();
    cur ^= 1;
  }

  // normalize and write ctx as [B][S][H*HD] f16 for the output GEMM
#pragma unroll
  for (int r = 0; r < 8; ++r) {
    float inv = 1.0f / lrow[r];
    int srow = qBase + w * 16 + lh * 8 + r;
    _Float16* op = ctxh + ((size_t)(b * SB + srow)) * (NH * HD) + h * HD;
#pragma unroll
    for (int d = 0; d < 8; ++d)
      op[d * 16 + lr] = (_Float16)(ctxAcc[d][r] * inv);
  }
}

// ---------------------------------------------------------------------------
extern "C" void kernel_launch(void* const* d_in, const int* in_sizes, int n_in,
                              void* d_out, int out_size, void* d_ws, size_t ws_size,
                              hipStream_t stream) {
  const float* x    = (const float*)d_in[0];
  // d_in[1] = mask (bool) — causal mask is computed analytically, unused.
  const float* cosb = (const float*)d_in[2];
  const float* sinb = (const float*)d_in[3];
  const float* Wq   = (const float*)d_in[4];
  const float* Wk   = (const float*)d_in[5];
  const float* Wv   = (const float*)d_in[6];
  const float* Wo   = (const float*)d_in[7];
  const float* qsc  = (const float*)d_in[8];
  const float* ksc  = (const float*)d_in[9];
  float* out = (float*)d_out;

  char* ws = (char*)d_ws;
  size_t off = 0;
  auto alloc = [&](size_t bytes) -> char* {
    char* p = ws + off;
    off += (bytes + 255) & ~(size_t)255;
    return p;
  };

  const size_t BS = (size_t)2 * SB;            // 4096 rows
  _Float16* xh    = (_Float16*)alloc(BS * SD * 2);
  _Float16* Wqh   = (_Float16*)alloc((size_t)SD * NH * HD * 2);
  _Float16* Wkh   = (_Float16*)alloc((size_t)SD * NKV * HD * 2);
  _Float16* Wvh   = (_Float16*)alloc((size_t)SD * NKV * HD * 2);
  _Float16* Woh   = (_Float16*)alloc((size_t)NH * HD * SD * 2);
  float*    q_raw = (float*)   alloc(BS * NH * HD * 4);
  float*    k_raw = (float*)   alloc(BS * NKV * HD * 4);
  float*    v_raw = (float*)   alloc(BS * NKV * HD * 4);
  _Float16* qh16  = (_Float16*)alloc(BS * NH * HD * 2);
  _Float16* kh16  = (_Float16*)alloc(BS * NKV * HD * 2);
  _Float16* vh16  = (_Float16*)alloc(BS * NKV * HD * 2);
  _Float16* ctxh  = (_Float16*)q_raw;          // reuse: q_raw dead after norm

  // fp32 -> f16
  cvt_f32_f16<<<4096, 256, 0, stream>>>(x,  xh,  BS * SD);
  cvt_f32_f16<<<4096, 256, 0, stream>>>(Wq, Wqh, (size_t)SD * NH * HD);
  cvt_f32_f16<<<2048, 256, 0, stream>>>(Wk, Wkh, (size_t)SD * NKV * HD);
  cvt_f32_f16<<<2048, 256, 0, stream>>>(Wv, Wvh, (size_t)SD * NKV * HD);
  cvt_f32_f16<<<4096, 256, 0, stream>>>(Wo, Woh, (size_t)NH * HD * SD);

  // projections (WMMA GEMMs)
  gemm_f16f32<<<dim3(16, 32), 256, 0, stream>>>(xh, Wqh, q_raw, 4096, 2048, 2048);
  gemm_f16f32<<<dim3(4,  32), 256, 0, stream>>>(xh, Wkh, k_raw, 4096,  512, 2048);
  gemm_f16f32<<<dim3(4,  32), 256, 0, stream>>>(xh, Wvh, v_raw, 4096,  512, 2048);

  // RMSNorm + RoPE (+ q scaling), transpose to head-major f16
  qk_norm_rope<<<2 * SB * NH,  128, 0, stream>>>(q_raw, qsc, cosb, sinb, qh16, NH,  SCALING);
  qk_norm_rope<<<2 * SB * NKV, 128, 0, stream>>>(k_raw, ksc, cosb, sinb, kh16, NKV, 1.0f);
  v_transpose<<<2048, 256, 0, stream>>>(v_raw, vh16);

  // flash attention (WMMA QK^T + PV, online softmax)
  attn_flash<<<dim3(2 * NH, SB / 128), 256, 0, stream>>>(qh16, kh16, vh16, ctxh);

  // output projection -> f32 d_out
  gemm_f16f32<<<dim3(16, 32), 256, 0, stream>>>(ctxh, Woh, out, 4096, 2048, 2048);
}
